// TTCL_36953898615116
// MI455X (gfx1250) — compile-verified, matrix operand order checked by reading, not run on hardware
//
#include <hip/hip_runtime.h>
#include <hip/hip_bf16.h>
#include <math.h>

typedef __attribute__((ext_vector_type(16))) _Float16 v16h;
typedef __attribute__((ext_vector_type(8)))  float    v8f;

#define TTCL_B   4096
#define TTCL_D   768
#define TTCL_INV_T (1.0f / 0.07f)
#define NEG_INF (-3.402823466e38f)

union Frag { v16h h; uint4 q[2]; };

// ---------------------------------------------------------------------------
// Kernel 1: row-normalize f32 -> f16.  One 256-thread block per row.
// blockIdx.x in [0, 2B): first B rows = text -> tH, next B = table -> zH.
// ---------------------------------------------------------------------------
__global__ __launch_bounds__(256)
void ttcl_normalize(const float* __restrict__ text, const float* __restrict__ table,
                    _Float16* __restrict__ tH, _Float16* __restrict__ zH) {
    const int  row     = blockIdx.x & (TTCL_B - 1);
    const bool isTable = blockIdx.x >= TTCL_B;
    const float*    src = (isTable ? table : text) + (size_t)row * TTCL_D;
    _Float16*       dst = (isTable ? zH    : tH  ) + (size_t)row * TTCL_D;

    __shared__ float red[256];
    float ss = 0.0f;
    for (int j = threadIdx.x; j < TTCL_D; j += 256) {
        float v = src[j];
        ss += v * v;
    }
    red[threadIdx.x] = ss;
    __syncthreads();
    for (int off = 128; off > 0; off >>= 1) {
        if (threadIdx.x < off) red[threadIdx.x] += red[threadIdx.x + off];
        __syncthreads();
    }
    const float inv = rsqrtf(red[0]);
    for (int j = threadIdx.x; j < TTCL_D; j += 256)
        dst[j] = (_Float16)(src[j] * inv);
}

// ---------------------------------------------------------------------------
// Kernel 2: sim = zH @ tH^T  via v_wmma_f32_16x16x32_f16 (wave32).
// Wave tile 64x64 (4x4 accumulators), 8 waves as 2(M)x4(N) -> block 128x256.
// Grid: (B/256, B/128).  NT layout: both operands read contiguous K rows.
// ---------------------------------------------------------------------------
__global__ __launch_bounds__(256)
void ttcl_gemm(const _Float16* __restrict__ zH, const _Float16* __restrict__ tH,
               float* __restrict__ sim) {
    const int lane   = threadIdx.x & 31;
    const int wave   = threadIdx.x >> 5;
    const int laneLo = lane & 15;
    const int sel    = lane >> 4;          // which half of the K layout

    const int mBase = blockIdx.y * 128 + (wave >> 2) * 64;  // table rows (z)
    const int nBase = blockIdx.x * 256 + (wave & 3) * 64;   // text rows (t)

    const _Float16* aRow[4];
    const _Float16* bRow[4];
#pragma unroll
    for (int i = 0; i < 4; ++i) {
        aRow[i] = zH + (size_t)(mBase + i * 16 + laneLo) * TTCL_D;
        bRow[i] = tH + (size_t)(nBase + i * 16 + laneLo) * TTCL_D;
    }

    v8f c[4][4];
#pragma unroll
    for (int mi = 0; mi < 4; ++mi)
#pragma unroll
        for (int ni = 0; ni < 4; ++ni)
            c[mi][ni] = (v8f){0.f, 0.f, 0.f, 0.f, 0.f, 0.f, 0.f, 0.f};

    for (int k0 = 0; k0 < TTCL_D; k0 += 32) {
        // 16-bit A-matrix lane layout (ISA 7.12.2): lanes 0-15 hold K=[0..7]
        // and K=[16..23]; lanes 16-31 hold K=[8..15] and K=[24..31].
        const int off0 = k0 + sel * 8;
        const int off1 = k0 + 16 + sel * 8;

        Frag a[4], b[4];
#pragma unroll
        for (int i = 0; i < 4; ++i) {
            a[i].q[0] = *(const uint4*)(aRow[i] + off0);
            a[i].q[1] = *(const uint4*)(aRow[i] + off1);
            b[i].q[0] = *(const uint4*)(bRow[i] + off0);
            b[i].q[1] = *(const uint4*)(bRow[i] + off1);
        }

#pragma unroll
        for (int mi = 0; mi < 4; ++mi)
#pragma unroll
            for (int ni = 0; ni < 4; ++ni)
                c[mi][ni] = __builtin_amdgcn_wmma_f32_16x16x32_f16(
                    /*neg_a=*/false, a[mi].h, /*neg_b=*/false, b[ni].h,
                    /*c_mod=*/(short)0, c[mi][ni],
                    /*reuse_a=*/false, /*reuse_b=*/false);
    }

    // C/D layout: VGPR r, lanes 0-15 -> M=r, N=lane; lanes 16-31 -> M=r+8.
#pragma unroll
    for (int mi = 0; mi < 4; ++mi) {
#pragma unroll
        for (int ni = 0; ni < 4; ++ni) {
            const int col = nBase + ni * 16 + laneLo;
#pragma unroll
            for (int r = 0; r < 8; ++r) {
                const int row = mBase + mi * 16 + r + 8 * sel;
                sim[(size_t)row * TTCL_B + col] = c[mi][ni][r];
            }
        }
    }
}

// ---------------------------------------------------------------------------
// Kernel 3: per-sample top-5 hard-negative loss, both directions.
// One 256-thread block per sample i. Row pass (z2t) + column pass (t2z).
// Deterministic: fixed-order tree argmax reductions, no atomics.
// ---------------------------------------------------------------------------
__global__ __launch_bounds__(256)
void ttcl_topk_loss(const float* __restrict__ sim, float* __restrict__ partial) {
    const int i   = blockIdx.x;
    const int tid = threadIdx.x;

    __shared__ float cand[256 * 5];
    __shared__ float redV[256];
    __shared__ int   redI[256];
    __shared__ float topS[5];

    const float posv  = sim[(size_t)i * TTCL_B + i] * TTCL_INV_T;
    float       total = 0.0f;

    for (int dir = 0; dir < 2; ++dir) {
        // per-thread sorted top-5 (t0 >= t1 >= ... >= t4)
        float t0 = NEG_INF, t1 = NEG_INF, t2 = NEG_INF, t3 = NEG_INF, t4 = NEG_INF;
        for (int j = tid; j < TTCL_B; j += 256) {
            float v = (dir == 0) ? sim[(size_t)i * TTCL_B + j]
                                 : sim[(size_t)j * TTCL_B + i];
            if (j == i) v = NEG_INF;
            if (v > t4) {
                if      (v > t0) { t4 = t3; t3 = t2; t2 = t1; t1 = t0; t0 = v; }
                else if (v > t1) { t4 = t3; t3 = t2; t2 = t1; t1 = v; }
                else if (v > t2) { t4 = t3; t3 = t2; t2 = v; }
                else if (v > t3) { t4 = t3; t3 = v; }
                else             { t4 = v; }
            }
        }
        cand[tid * 5 + 0] = t0; cand[tid * 5 + 1] = t1; cand[tid * 5 + 2] = t2;
        cand[tid * 5 + 3] = t3; cand[tid * 5 + 4] = t4;
        __syncthreads();

        // 5 rounds of global argmax over the 1280 candidates
        for (int s = 0; s < 5; ++s) {
            float mv = cand[tid * 5];
            int   mi = tid * 5;
#pragma unroll
            for (int q = 1; q < 5; ++q) {
                float v = cand[tid * 5 + q];
                if (v > mv) { mv = v; mi = tid * 5 + q; }
            }
            redV[tid] = mv; redI[tid] = mi;
            __syncthreads();
            for (int off = 128; off > 0; off >>= 1) {
                if (tid < off && redV[tid + off] > redV[tid]) {
                    redV[tid] = redV[tid + off];
                    redI[tid] = redI[tid + off];
                }
                __syncthreads();
            }
            if (tid == 0) { topS[s] = redV[0]; cand[redI[0]] = NEG_INF; }
            __syncthreads();
        }

        if (tid == 0) {
            float l[6];
            l[0] = posv;
#pragma unroll
            for (int q = 0; q < 5; ++q) l[q + 1] = topS[q] * TTCL_INV_T;
            float m = l[0];
#pragma unroll
            for (int q = 1; q < 6; ++q) m = fmaxf(m, l[q]);
            float se = 0.0f;
#pragma unroll
            for (int q = 0; q < 6; ++q) se += expf(l[q] - m);
            total += (m + logf(se)) - posv;   // -log_softmax[0]
        }
        __syncthreads();
    }
    if (tid == 0) partial[i] = total;
}

// ---------------------------------------------------------------------------
// Kernel 4: fixed-order final reduction -> loss scalar.
// ---------------------------------------------------------------------------
__global__ __launch_bounds__(256)
void ttcl_reduce(const float* __restrict__ partial, float* __restrict__ out) {
    __shared__ float red[256];
    float acc = 0.0f;
    for (int j = threadIdx.x; j < TTCL_B; j += 256) acc += partial[j];
    red[threadIdx.x] = acc;
    __syncthreads();
    for (int off = 128; off > 0; off >>= 1) {
        if (threadIdx.x < off) red[threadIdx.x] += red[threadIdx.x + off];
        __syncthreads();
    }
    if (threadIdx.x == 0) out[0] = red[0] / (2.0f * TTCL_B);
}

// ---------------------------------------------------------------------------
extern "C" void kernel_launch(void* const* d_in, const int* in_sizes, int n_in,
                              void* d_out, int out_size, void* d_ws, size_t ws_size,
                              hipStream_t stream) {
    const float* text  = (const float*)d_in[0];   // text_emb  [B, D] f32
    const float* table = (const float*)d_in[1];   // table_emb [B, D] f32

    float* out = (float*)d_out;     // out[0] = loss; out[1..] = sim [B, B]
    float* sim = out + 1;

    char* ws = (char*)d_ws;
    _Float16* tH      = (_Float16*)ws;                                   // 6.29 MB
    _Float16* zH      = (_Float16*)(ws + (size_t)TTCL_B * TTCL_D * 2);   // 6.29 MB
    float*    partial = (float*)  (ws + (size_t)TTCL_B * TTCL_D * 4);    // 16 KB

    ttcl_normalize<<<2 * TTCL_B, 256, 0, stream>>>(text, table, tH, zH);

    dim3 grid(TTCL_B / 256, TTCL_B / 128);
    ttcl_gemm<<<grid, 256, 0, stream>>>(zH, tH, sim);

    ttcl_topk_loss<<<TTCL_B, 256, 0, stream>>>(sim, partial);
    ttcl_reduce<<<1, 256, 0, stream>>>(partial, out);
}
